// InterpretableMultiHeadAttention_44238163149183
// MI455X (gfx1250) — matrix operand order repaired and checked by reading, not compiled, not run on hardware
//
#include <hip/hip_runtime.h>

// ---------------------------------------------------------------------------
// InterpretableMultiHeadAttention for MI455X (gfx1250, wave32, WMMA).
// B=4, S=2048, H=1024, 16 heads x d=64.
// d_out = [ output (B*S*H f32) | avg_attn (B*S*S f32) ]
// ---------------------------------------------------------------------------

#define B_  4
#define S_  2048
#define H_  1024
#define NH_ 16
#define D_  64

typedef unsigned short u16;
typedef unsigned int   u32;

typedef __bf16 v16bf __attribute__((ext_vector_type(16)));
typedef float  v8f   __attribute__((ext_vector_type(8)));
typedef float  v2f   __attribute__((ext_vector_type(2)));

union FragBF { v16bf v; uint4 q[2]; };   // 32B: one 16x32 A-frag / 32x16 B-frag slice per lane

#if defined(__has_builtin)
#if __has_builtin(__builtin_amdgcn_global_load_async_to_lds_b128)
#define ASYNC_LDS 1
#endif
#endif

#ifdef ASYNC_LDS
// builtin signature (from compiler diagnostic): first param is a pointer to a
// 16B int vector in the global address space; second in LDS address space.
typedef int v4i_ __attribute__((__vector_size__(16)));
typedef __attribute__((address_space(1))) v4i_ gv4i;
typedef __attribute__((address_space(3))) v4i_ lv4i;
#endif

__device__ __forceinline__ u16 f2bf(float f) {
  union { float f; u32 u; } x; x.f = f;
  u32 u = x.u;
  u32 r = (u + 0x7FFFu + ((u >> 16) & 1u)) >> 16;   // round-to-nearest-even
  return (u16)r;
}

__device__ __forceinline__ float redmax16(float v) {
  v = fmaxf(v, __shfl_xor(v, 8, 32));
  v = fmaxf(v, __shfl_xor(v, 4, 32));
  v = fmaxf(v, __shfl_xor(v, 2, 32));
  v = fmaxf(v, __shfl_xor(v, 1, 32));
  return v;   // xor masks <16 keep the reduction inside each 16-lane half
}
__device__ __forceinline__ float redsum16(float v) {
  v += __shfl_xor(v, 8, 32);
  v += __shfl_xor(v, 4, 32);
  v += __shfl_xor(v, 2, 32);
  v += __shfl_xor(v, 1, 32);
  return v;
}

// ---------------------------------------------------------------------------
// fp32 -> bf16 conversion (vector-of-4 per thread)
// ---------------------------------------------------------------------------
__global__ void cvt_f32_bf16(const float* __restrict__ src, u16* __restrict__ dst, int n) {
  int i = (blockIdx.x * blockDim.x + threadIdx.x) * 4;
  if (i + 3 < n) {
    float4 f = *(const float4*)(src + i);
    u32 lo = (u32)f2bf(f.x) | ((u32)f2bf(f.y) << 16);
    u32 hi = (u32)f2bf(f.z) | ((u32)f2bf(f.w) << 16);
    *(uint2*)(dst + i) = make_uint2(lo, hi);
  }
}

// ---------------------------------------------------------------------------
// C[M,N] = A[M,K](bf16) @ W[N,K](bf16)^T + bias[N], f32 accumulate.
// Register-blocked: one wave computes a 32x32 block = 2x2 WMMA tiles
// (4 WMMAs per 2 A-frags + 2 B-frags -> half the fragment bytes per FLOP).
// 4 waves/block; BWN waves along N, 4/BWN along M.
//   block tile: (4/BWN)*32 rows x BWN*32 cols, grid=(N/(BWN*32), M/((4/BWN)*32))
// ---------------------------------------------------------------------------
template<int BWN, bool BF16_OUT>
__global__ void __launch_bounds__(128)
gemm_wmma(const u16* __restrict__ A, const u16* __restrict__ W,
          const float* __restrict__ bias, void* __restrict__ Cv,
          int M, int N, int K) {
  const int lane = threadIdx.x & 31;
  const int wave = threadIdx.x >> 5;
  const int half = lane >> 4;
  const int r    = lane & 15;
  const int wn = wave % BWN;
  const int wm = wave / BWN;
  const int n0 = blockIdx.x * (BWN * 32) + wn * 32;
  const int m0 = blockIdx.y * ((4 / BWN) * 32) + wm * 32;

  const u16* Arow0 = A + (size_t)(m0 + r) * K;
  const u16* Arow1 = Arow0 + (size_t)16 * K;
  const u16* Wrow0 = W + (size_t)(n0 + r) * K;
  const u16* Wrow1 = Wrow0 + (size_t)16 * K;

  v8f acc00 = {}, acc01 = {}, acc10 = {}, acc11 = {};
  for (int k0 = 0; k0 < K; k0 += 32) {
    FragBF a0, a1, b0, b1;
    a0.q[0] = *(const uint4*)(Arow0 + k0 + 8 * half);
    a0.q[1] = *(const uint4*)(Arow0 + k0 + 16 + 8 * half);
    a1.q[0] = *(const uint4*)(Arow1 + k0 + 8 * half);
    a1.q[1] = *(const uint4*)(Arow1 + k0 + 16 + 8 * half);
    b0.q[0] = *(const uint4*)(Wrow0 + k0 + 8 * half);
    b0.q[1] = *(const uint4*)(Wrow0 + k0 + 16 + 8 * half);
    b1.q[0] = *(const uint4*)(Wrow1 + k0 + 8 * half);
    b1.q[1] = *(const uint4*)(Wrow1 + k0 + 16 + 8 * half);
    __builtin_prefetch(Arow0 + k0 + 32, 0, 0);   // global_prefetch_b8
    acc00 = __builtin_amdgcn_wmma_f32_16x16x32_bf16(false, a0.v, false, b0.v, (short)0, acc00, false, false);
    acc01 = __builtin_amdgcn_wmma_f32_16x16x32_bf16(false, a0.v, false, b1.v, (short)0, acc01, false, false);
    acc10 = __builtin_amdgcn_wmma_f32_16x16x32_bf16(false, a1.v, false, b0.v, (short)0, acc10, false, false);
    acc11 = __builtin_amdgcn_wmma_f32_16x16x32_bf16(false, a1.v, false, b1.v, (short)0, acc11, false, false);
  }
  const float bc0 = bias[n0 + r];
  const float bc1 = bias[n0 + 16 + r];
#pragma unroll
  for (int i = 0; i < 8; ++i) {
    const int row0 = m0 + i + 8 * half;
    const int row1 = row0 + 16;
    if (BF16_OUT) {
      u16* C = (u16*)Cv;
      C[(size_t)row0 * N + n0 + r]      = f2bf(acc00[i] + bc0);
      C[(size_t)row0 * N + n0 + 16 + r] = f2bf(acc01[i] + bc1);
      C[(size_t)row1 * N + n0 + r]      = f2bf(acc10[i] + bc0);
      C[(size_t)row1 * N + n0 + 16 + r] = f2bf(acc11[i] + bc1);
    } else {
      float* C = (float*)Cv;
      C[(size_t)row0 * N + n0 + r]      = acc00[i] + bc0;
      C[(size_t)row0 * N + n0 + 16 + r] = acc01[i] + bc1;
      C[(size_t)row1 * N + n0 + r]      = acc10[i] + bc0;
      C[(size_t)row1 * N + n0 + 16 + r] = acc11[i] + bc1;
    }
  }
}

// ---------------------------------------------------------------------------
// Attention-weights kernel: avg[b,q,k] = (1/16) * sum_h softmax_k(Q_h.K_h/8)
// for one 16-row query tile per block. blockDim=512 (16 waves), grid=B*S/16.
//  * Q row-block (16x1024 bf16, all heads = 32 KB) staged in LDS once via
//    async global->LDS loads; both passes read A-frags with ds_load_b128.
//  * Pass 1: wave w == head w streams 128 col tiles keeping PER-LANE flash
//    (max,sumexp) -- no cross-lane ops in the loop; one 4-step shuffle merge
//    at the end. Stats -> 2 KB LDS.
//  * Pass 2: wave w owns col tiles {w, w+16,...}; loops all 16 heads
//    accumulating normalized weights in registers; stores straight to global.
// ---------------------------------------------------------------------------
__global__ void __launch_bounds__(512)
attn_avg_kernel(const u16* __restrict__ Qb, const u16* __restrict__ Kb,
                float* __restrict__ avg) {
  __shared__ u16   sQ[16][H_];     // 32 KB: 16 query rows x all heads
  __shared__ float sm[NH_][16];
  __shared__ float sl[NH_][16];

  const int lane = threadIdx.x & 31;
  const int wave = threadIdx.x >> 5;   // 0..15
  const int half = lane >> 4;
  const int r    = lane & 15;

  const int b  = blockIdx.x >> 7;           // S/16 = 128 q-tiles per batch
  const int q0 = (blockIdx.x & 127) * 16;

  const u16* Kbase = Kb + (size_t)b * S_ * H_;

  // ---- stage Q tile into LDS (async DMA if available) ----
  {
    const int tid = threadIdx.x;
#pragma unroll
    for (int j = 0; j < 4; ++j) {
      const int cid = tid + 512 * j;        // 2048 x 16B chunks
      const int row = cid >> 7;
      const int cc  = cid & 127;
      const u16* g = Qb + ((size_t)b * S_ + q0 + row) * H_ + cc * 8;
#ifdef ASYNC_LDS
      // generic LDS pointers carry the byte offset in addr[31:0] (ISA 10.2),
      // so integer casts give valid AS1/AS3 pointers without addrspacecast.
      __builtin_amdgcn_global_load_async_to_lds_b128(
          (gv4i*)(unsigned long long)g,
          (lv4i*)(unsigned int)(size_t)&sQ[row][cc * 8], 0, 0);
#else
      *(uint4*)&sQ[row][cc * 8] = *(const uint4*)g;
#endif
    }
#ifdef ASYNC_LDS
#if __has_builtin(__builtin_amdgcn_s_wait_asynccnt)
    __builtin_amdgcn_s_wait_asynccnt(0);
#else
    asm volatile("s_wait_asynccnt 0" ::: "memory");
#endif
#endif
    __syncthreads();
  }

  // ---------------- pass 1: stats for head h = wave ----------------
  {
    const int h = wave;
    const u16* Qh = &sQ[r][h * D_];
    float m8[8], l8[8];
#pragma unroll
    for (int i = 0; i < 8; ++i) { m8[i] = -1e30f; l8[i] = 0.0f; }

    for (int c0 = 0; c0 < S_; c0 += 16) {
      const u16* Kh = Kbase + (size_t)(c0 + r) * H_ + h * D_;
      v8f c = {};
#pragma unroll
      for (int k0 = 0; k0 < D_; k0 += 32) {
        FragBF a, bb;
        a.q[0]  = *(const uint4*)(Qh + k0 + 8 * half);       // ds_load_b128
        a.q[1]  = *(const uint4*)(Qh + k0 + 16 + 8 * half);
        bb.q[0] = *(const uint4*)(Kh + k0 + 8 * half);
        bb.q[1] = *(const uint4*)(Kh + k0 + 16 + 8 * half);
        c = __builtin_amdgcn_wmma_f32_16x16x32_bf16(
            false, a.v, false, bb.v, (short)0, c, false, false);
      }
      // per-lane online softmax update: each lane owns one column per row
#pragma unroll
      for (int i = 0; i < 8; ++i) {
        const float s  = c[i] * 0.125f;          // HEAD_DIM^-0.5
        const float nm = fmaxf(m8[i], s);
        l8[i] = l8[i] * __expf(m8[i] - nm) + __expf(s - nm);
        m8[i] = nm;
      }
    }
    // single cross-lane merge over the 16 columns owned by this half
#pragma unroll
    for (int i = 0; i < 8; ++i) {
      const float M = redmax16(m8[i]);
      const float L = redsum16(l8[i] * __expf(m8[i] - M));
      m8[i] = M; l8[i] = L;
    }
    if (r == 0) {
#pragma unroll
      for (int i = 0; i < 8; ++i) {
        sm[h][i + 8 * half] = m8[i];
        sl[h][i + 8 * half] = l8[i];
      }
    }
  }
  __syncthreads();

  // ---------------- pass 2: normalize + head-average ----------------
  for (int t = 0; t < 8; ++t) {
    const int c0 = (wave + 16 * t) * 16;
    v8f accum = {};
    for (int h = 0; h < NH_; ++h) {
      const u16* Qh = &sQ[r][h * D_];
      const u16* Kh = Kbase + (size_t)(c0 + r) * H_ + h * D_;
      v8f c = {};
#pragma unroll
      for (int k0 = 0; k0 < D_; k0 += 32) {
        FragBF a, bb;
        a.q[0]  = *(const uint4*)(Qh + k0 + 8 * half);
        a.q[1]  = *(const uint4*)(Qh + k0 + 16 + 8 * half);
        bb.q[0] = *(const uint4*)(Kh + k0 + 8 * half);
        bb.q[1] = *(const uint4*)(Kh + k0 + 16 + 8 * half);
        c = __builtin_amdgcn_wmma_f32_16x16x32_bf16(
            false, a.v, false, bb.v, (short)0, c, false, false);
      }
#pragma unroll
      for (int i = 0; i < 8; ++i) {
        const int row = i + 8 * half;
        accum[i] += __expf(c[i] * 0.125f - sm[h][row]) * (0.0625f / sl[h][row]);
      }
    }
    float* out = avg + ((size_t)b * S_ + q0) * S_ + c0;
#pragma unroll
    for (int i = 0; i < 8; ++i)
      out[(size_t)(i + 8 * half) * S_ + r] = accum[i];
  }
}

// ---------------------------------------------------------------------------
// attn_out(bf16) = avg(f32)[b,S,S] @ V(f32)[b,S,64] via V_WMMA_F32_16X16X4_F32.
// One wave = one 16x16 tile; dual accumulators break the WMMA RAW chain.
// ---------------------------------------------------------------------------
__global__ void __launch_bounds__(256)
av_wmma_f32(const float* __restrict__ avg, const float* __restrict__ V,
            u16* __restrict__ Obf) {
  const int lane = threadIdx.x & 31;
  const int wave = threadIdx.x >> 5;   // 0..7
  const int half = lane >> 4;
  const int r    = lane & 15;
  const int brow0 = blockIdx.x * 32 + (wave >> 2) * 16;  // row in [0,8192)
  const int n0    = (wave & 3) * 16;
  const int b = brow0 >> 11;           // /2048
  const int q = brow0 & (S_ - 1);

  const float* Arow = avg + ((size_t)b * S_ + q + r) * S_;
  const float* Vb   = V + (size_t)b * S_ * D_;

  v8f acc0 = {}, acc1 = {};
  for (int k0 = 0; k0 < S_; k0 += 8) {
    v2f a0 = *(const v2f*)(Arow + k0 + 2 * half);
    v2f a1 = *(const v2f*)(Arow + k0 + 4 + 2 * half);
    v2f b0, b1;
    b0[0] = Vb[(size_t)(k0 + 2 * half) * D_ + n0 + r];
    b0[1] = Vb[(size_t)(k0 + 2 * half + 1) * D_ + n0 + r];
    b1[0] = Vb[(size_t)(k0 + 4 + 2 * half) * D_ + n0 + r];
    b1[1] = Vb[(size_t)(k0 + 4 + 2 * half + 1) * D_ + n0 + r];
    acc0 = __builtin_amdgcn_wmma_f32_16x16x4_f32(false, a0, false, b0, (short)0, acc0, false, false);
    acc1 = __builtin_amdgcn_wmma_f32_16x16x4_f32(false, a1, false, b1, (short)0, acc1, false, false);
  }
  v8f acc = acc0 + acc1;
#pragma unroll
  for (int i = 0; i < 8; ++i)
    Obf[(size_t)(brow0 + i + 8 * half) * D_ + n0 + r] = f2bf(acc[i]);
}

// ---------------------------------------------------------------------------
// Host launcher
// ---------------------------------------------------------------------------
extern "C" void kernel_launch(void* const* d_in, const int* in_sizes, int n_in,
                              void* d_out, int out_size, void* d_ws, size_t ws_size,
                              hipStream_t stream) {
  (void)in_sizes; (void)n_in; (void)out_size; (void)ws_size;

  const float* query  = (const float*)d_in[0];
  const float* key_in = (const float*)d_in[1];
  const float* value  = (const float*)d_in[2];
  const float* Wq = (const float*)d_in[3];
  const float* bq = (const float*)d_in[4];
  const float* Wk = (const float*)d_in[5];
  const float* bk = (const float*)d_in[6];
  const float* Wv = (const float*)d_in[7];
  const float* bv = (const float*)d_in[8];
  const float* Wo = (const float*)d_in[9];
  const float* bo = (const float*)d_in[10];

  float* out = (float*)d_out;                         // [B,S,H]
  float* avg = out + (size_t)B_ * S_ * H_;            // [B,S,S]

  char* wsp = (char*)d_ws;
  auto alloc = [&](size_t bytes) -> char* {
    char* p = wsp; wsp += (bytes + 255) & ~(size_t)255; return p;
  };
  const size_t BSH = (size_t)B_ * S_ * H_;            // 8388608
  u16* qin_bf = (u16*)alloc(BSH * 2);
  u16* kin_bf = (u16*)alloc(BSH * 2);
  u16* vin_bf = (u16*)alloc(BSH * 2);
  u16* wq_bf  = (u16*)alloc((size_t)H_ * H_ * 2);
  u16* wk_bf  = (u16*)alloc((size_t)H_ * H_ * 2);
  u16* wv_bf  = (u16*)alloc((size_t)D_ * H_ * 2);
  u16* wo_bf  = (u16*)alloc((size_t)H_ * D_ * 2);
  u16* Qbf    = (u16*)alloc(BSH * 2);
  u16* Kbf    = (u16*)alloc(BSH * 2);
  float* Vf   = (float*)alloc((size_t)B_ * S_ * D_ * 4);
  u16* AObf   = (u16*)alloc((size_t)B_ * S_ * D_ * 2);

  auto cvt = [&](const float* s, u16* d, size_t n) {
    cvt_f32_bf16<<<dim3((unsigned)(n / 1024)), dim3(256), 0, stream>>>(s, d, (int)n);
  };
  cvt(query,  qin_bf, BSH);
  cvt(key_in, kin_bf, BSH);
  cvt(value,  vin_bf, BSH);
  cvt(Wq, wq_bf, (size_t)H_ * H_);
  cvt(Wk, wk_bf, (size_t)H_ * H_);
  cvt(Wv, wv_bf, (size_t)D_ * H_);
  cvt(Wo, wo_bf, (size_t)H_ * D_);

  const int M = B_ * S_;  // 8192
  // Q = query @ Wq^T + bq  (bf16 out): block 32x128, grid (8, 256)
  gemm_wmma<4, true><<<dim3(H_ / 128, M / 32), dim3(128), 0, stream>>>(
      qin_bf, wq_bf, bq, Qbf, M, H_, H_);
  // K = key_in @ Wk^T + bk (bf16 out)
  gemm_wmma<4, true><<<dim3(H_ / 128, M / 32), dim3(128), 0, stream>>>(
      kin_bf, wk_bf, bk, Kbf, M, H_, H_);
  // V = value @ Wv^T + bv  (f32 out, [B*S, 64]): block 64x64, grid (1, 128)
  gemm_wmma<2, false><<<dim3(D_ / 64, M / 64), dim3(128), 0, stream>>>(
      vin_bf, wv_bf, bv, Vf, M, D_, H_);

  // head-averaged softmax weights -> avg (second half of d_out)
  attn_avg_kernel<<<dim3(B_ * (S_ / 16)), dim3(512), 0, stream>>>(Qbf, Kbf, avg);

  // attn_out = avg @ V (f32 WMMA, bf16 out)
  av_wmma_f32<<<dim3(M / 32), dim3(256), 0, stream>>>(avg, Vf, AObf);

  // output = attn_out @ Wo^T + bo (f32 out -> first half of d_out)
  gemm_wmma<4, false><<<dim3(H_ / 128, M / 32), dim3(128), 0, stream>>>(
      AObf, wo_bf, bo, out, M, H_, 64);
}
